// Transfomer_171798692112
// MI455X (gfx1250) — matrix operand order repaired and checked by reading, not compiled, not run on hardware
//
#include <hip/hip_runtime.h>

// ---------------------------------------------------------------------------
// Problem constants (fixed by the reference)
// ---------------------------------------------------------------------------
#define T_B   4
#define T_S   512
#define T_D   512
#define T_DR  64
#define T_H   8
#define T_M   (T_B * T_S)     // 2048 tokens total
#define T_DFF (4 * T_D)       // 2048
#define T_V   32000
#define LN_EPS 1e-3f

typedef __attribute__((ext_vector_type(16))) _Float16 v16h;
typedef __attribute__((ext_vector_type(8)))  float    v8f;
typedef __attribute__((ext_vector_type(4)))  unsigned int v4u;
typedef __attribute__((ext_vector_type(8)))  int      v8i;
typedef __attribute__((ext_vector_type(4)))  int      v4i;

// ---------------------------------------------------------------------------
// WMMA GEMM:  C[m,n] = (sum_k A[m,k] * Bt[n,k] + bias[n]) * scale  (opt. ReLU)
//   A  : f16 row-major [M,K]   (lda elements per row)
//   Bt : f16 row-major [N,K]   (i.e. B transposed; ldb elements per row)
//   C  : f32 row-major [M,N]
// Batched via blockIdx.z with decomposed offsets:
//   off = (z / zdiv) * strideOuter + (z % zdiv) * strideInner
// Tile staging: Tensor Data Mover (tensor_load_to_lds) for full tiles,
// guarded global->VGPR->LDS path for edge tiles.
// ---------------------------------------------------------------------------
#define BM 128
#define BN 128
#define BK 32
#define LDSS 40   // LDS row stride (halves): 80B = 64B data + 16B pad, 16B-aligned

union FragU { uint4 u[2]; v16h v; };

// One TDM descriptor-driven load of a BMxBK (128x32) f16 tile into LDS with
// 16B padding after each 64B row (compacted row stride = 80B = LDSS halves).
// D# layout per CDNA5 ISA 8.3/8.4. 2-D tensor: groups 2/3 zero.
// This toolchain exposes the 6-arg builtin:
//   (uint32x4 g0, int32x8 g1, int32x4 g2, int32x4 g3, int32x8 extra, i32 cpol)
__device__ __forceinline__ void tdm_load_tile(const _Float16* gptr, long ldElems,
                                              unsigned ldsAddr)
{
  unsigned long long ga = (unsigned long long)(uintptr_t)gptr;   // byte address
  v4u g0 = { 1u,                                   // count=1, user desc
             ldsAddr,                              // lds_addr [63:32]
             (unsigned)ga,                         // global_addr [95:64]
             (unsigned)((ga >> 32) & 0x01FFFFFFu) | (2u << 30) }; // addr hi + type=2
  unsigned long long st = (unsigned long long)ldElems;            // dim0 stride (elems)
  v8i g1 = { (int)((1u << 16)      // data_size = 1 -> 2 bytes (f16)
                 | (1u << 20)      // pad_enable
                 | (3u << 22)      // pad_interval: 16 DWORDs (64B)
                 | (3u << 25)),    // pad_amount: 4 DWORDs (16B)
             (int)(32u << 16),                     // tensor_dim0 = 32 (lo16 @ bits63:48)
             (int)(128u << 16),                    // td0 hi=0 | tensor_dim1 = 128 lo16
             (int)(32u << 16),                     // td1 hi=0 | tile_dim0 = 32
             (int)128,                             // tile_dim1 = 128, tile_dim2 = 0
             (int)(unsigned)(st & 0xffffffffu),    // tensor_dim0_stride lo32
             (int)(unsigned)((st >> 32) & 0xffffu),// stride hi16 | dim1_stride lo16 = 0
             0 };
  v4i gz = { 0, 0, 0, 0 };
  v8i gz8 = { 0, 0, 0, 0, 0, 0, 0, 0 };
  __builtin_amdgcn_tensor_load_to_lds(g0, g1, gz, gz, gz8, 0);
}

__global__ __launch_bounds__(256) void gemm_f16_wmma(
    const _Float16* __restrict__ A, long lda,
    const _Float16* __restrict__ Bt, long ldb,
    float* __restrict__ C, long ldc,
    int M, int N, int K,
    const float* __restrict__ bias, float scale, int relu,
    int zdiv, long aO, long aI, long bO, long bI, long cO, long cI)
{
  __shared__ __align__(16) _Float16 As[2][BM * LDSS];
  __shared__ __align__(16) _Float16 Bs[2][BN * LDSS];

  const int tid  = threadIdx.x;
  const int lane = tid & 31;
  const int w    = tid >> 5;
  const int wm   = w & 3;     // 4 strips of 32 rows
  const int wn   = w >> 2;    // 2 halves of 64 cols
  const int z    = blockIdx.z;
  const long zo  = z / zdiv;
  const long zi  = z % zdiv;

  const _Float16* Ab = A  + zo * aO + zi * aI;
  const _Float16* Bb = Bt + zo * bO + zi * bI;
  float*          Cb = C  + zo * cO + zi * cI;

  const int m0 = blockIdx.y * BM;
  const int n0 = blockIdx.x * BN;

  // Uniform per block: full tile -> single-instruction TDM staging.
  const bool useTDM = (m0 + BM <= M) && (n0 + BN <= N) && ((K & (BK - 1)) == 0);

  v8f zacc = {0.f, 0.f, 0.f, 0.f, 0.f, 0.f, 0.f, 0.f};
  v8f acc[2][4];
#pragma unroll
  for (int i = 0; i < 2; ++i)
#pragma unroll
    for (int j = 0; j < 4; ++j) acc[i][j] = zacc;

  auto stage = [&](int buf, int k0) {   // guarded fallback path (edge tiles)
#pragma unroll
    for (int i = 0; i < 2; ++i) {
      int idx = i * 256 + tid;          // 0..511
      int row = idx >> 2;               // 0..127
      int kc  = (idx & 3) << 3;         // 0,8,16,24
      int gk  = k0 + kc;
      uint4 va = make_uint4(0u, 0u, 0u, 0u);
      int gm = m0 + row;
      if (gm < M && gk < K) va = *(const uint4*)(Ab + (long)gm * lda + gk);
      *(uint4*)&As[buf][row * LDSS + kc] = va;
      uint4 vb = make_uint4(0u, 0u, 0u, 0u);
      int gn = n0 + row;
      if (gn < N && gk < K) vb = *(const uint4*)(Bb + (long)gn * ldb + gk);
      *(uint4*)&Bs[buf][row * LDSS + kc] = vb;
    }
  };
  auto tdm_stage = [&](int buf, int k0) {  // one wave issues 2 TDM tile loads
    tdm_load_tile(Ab + (long)m0 * lda + k0, lda,
                  (unsigned)(uintptr_t)&As[buf][0]);
    tdm_load_tile(Bb + (long)n0 * ldb + k0, ldb,
                  (unsigned)(uintptr_t)&Bs[buf][0]);
  };

  const int  mrow = lane & 15;
  const bool hi   = lane >= 16;
  const int  nT   = (K + BK - 1) / BK;

  if (useTDM) {
    if (w == 0) {
      tdm_stage(0, 0);
      __builtin_amdgcn_s_wait_tensorcnt(0);
    }
  } else {
    stage(0, 0);
  }
  __syncthreads();

  for (int kt = 0; kt < nT; ++kt) {
    int buf = kt & 1;
    if (kt + 1 < nT) {
      if (useTDM) {
        if (w == 0) tdm_stage(buf ^ 1, (kt + 1) * BK);
      } else {
        stage(buf ^ 1, (kt + 1) * BK);
        if (kt + 2 < nT) {  // stream the tile after next into GL2
          __builtin_prefetch(Ab + (long)(m0 + (tid >> 1)) * lda + (kt + 2) * BK, 0, 1);
          __builtin_prefetch(Bb + (long)(n0 + (tid >> 1)) * ldb + (kt + 2) * BK, 0, 1);
        }
      }
    }

    v16h af[2], bfv[4];
#pragma unroll
    for (int fm = 0; fm < 2; ++fm) {
      // A 16x32 f16 layout: lanes 0-15 -> K 0..7 & 16..23 ; lanes 16-31 -> K 8..15 & 24..31
      int r = wm * 32 + fm * 16 + mrow;
      const _Float16* p = &As[buf][r * LDSS];
      FragU f;
      f.u[0] = *(const uint4*)(p + (hi ? 8 : 0));
      f.u[1] = *(const uint4*)(p + (hi ? 24 : 16));
      af[fm] = f.v;
    }
#pragma unroll
    for (int fn = 0; fn < 4; ++fn) {
      // B 32x16 f16 layout: lanes 0-15 -> K 0..7 & 16..23 ; lanes 16-31 -> K 8..15 & 24..31
      int c = wn * 64 + fn * 16 + mrow;
      const _Float16* p = &Bs[buf][c * LDSS];
      FragU f;
      f.u[0] = *(const uint4*)(p + (hi ? 8 : 0));
      f.u[1] = *(const uint4*)(p + (hi ? 24 : 16));
      bfv[fn] = f.v;
    }
#pragma unroll
    for (int fm = 0; fm < 2; ++fm)
#pragma unroll
      for (int fn = 0; fn < 4; ++fn)
        acc[fm][fn] = __builtin_amdgcn_wmma_f32_16x16x32_f16(
            false, af[fm], false, bfv[fn], (short)0, acc[fm][fn], false, false);

    if (useTDM && w == 0) __builtin_amdgcn_s_wait_tensorcnt(0);
    __syncthreads();
  }

  // C layout: VGPR j -> (lane<16: M=j, lane>=16: M=8+j), N = lane&15
#pragma unroll
  for (int fm = 0; fm < 2; ++fm) {
    int rbase = m0 + wm * 32 + fm * 16 + (hi ? 8 : 0);
#pragma unroll
    for (int fn = 0; fn < 4; ++fn) {
      int col = n0 + wn * 64 + fn * 16 + mrow;
      if (col < N) {
        float bv = bias ? bias[col] : 0.0f;
#pragma unroll
        for (int j = 0; j < 8; ++j) {
          int row = rbase + j;
          if (row < M) {
            float v = (acc[fm][fn][j] + bv) * scale;
            if (relu) v = fmaxf(v, 0.0f);
            Cb[(long)row * ldc + col] = v;
          }
        }
      }
    }
  }
}

// ---------------------------------------------------------------------------
// Elementwise / packing / normalization kernels
// ---------------------------------------------------------------------------
__global__ void embed_kernel(const long long* __restrict__ idx,
                             const float* __restrict__ emb,
                             const float* __restrict__ pos,
                             float* __restrict__ out)
{
  int row = blockIdx.x;                              // 0..T_M-1
  long t = (long)idx[row];
  const float* e = emb + t * (long)T_D;
  const float* p = pos + (long)(row % T_S) * T_D;
  float* o = out + (long)row * T_D;
  for (int c = threadIdx.x; c < T_D; c += blockDim.x) o[c] = e[c] + p[c];
}

__global__ void cvt_f32_f16_kernel(const float* __restrict__ in,
                                   _Float16* __restrict__ out, long n)
{
  long i = (long)blockIdx.x * blockDim.x + threadIdx.x;
  if (i < n) out[i] = (_Float16)in[i];
}

// out[n*K + k] = in[k*N + n]   (pack a row-major [K,N] f32 weight into f16 [N,K])
__global__ void wt_transpose_kernel(const float* __restrict__ in,
                                    _Float16* __restrict__ out, int K, int N)
{
  long i = (long)blockIdx.x * blockDim.x + threadIdx.x;
  long total = (long)K * N;
  if (i < total) {
    long n = i / K, k = i % K;
    out[i] = (_Float16)in[k * (long)N + n];
  }
}

// Wq/Wk (H,D,DR) -> f16 [H*DR, D]:  out[c*D + d] = in[(c/DR)*D*DR + d*DR + (c%DR)]
__global__ void wt_heads_kernel(const float* __restrict__ in,
                                _Float16* __restrict__ out)
{
  long i = (long)blockIdx.x * blockDim.x + threadIdx.x;
  long total = (long)T_H * T_DR * T_D;
  if (i < total) {
    long c = i / T_D, d = i % T_D;
    long h = c >> 6, r = c & 63;  // DR = 64
    out[i] = (_Float16)in[h * (long)(T_D * T_DR) + d * T_DR + r];
  }
}

// V^T pack: out[((b*H+h)*DR + r)*S + sk] = kf[(b*S+sk)*(H*DR) + h*DR + r]
__global__ void vt_pack_kernel(const float* __restrict__ kf,
                               _Float16* __restrict__ out)
{
  long i = (long)blockIdx.x * blockDim.x + threadIdx.x;
  long total = (long)T_B * T_H * T_DR * T_S;
  if (i < total) {
    long sk = i % T_S; long t = i / T_S;
    long r = t % T_DR; t /= T_DR;
    long h = t % T_H;  long b = t / T_H;
    out[i] = (_Float16)kf[(b * T_S + sk) * (long)(T_H * T_DR) + h * T_DR + r];
  }
}

// LayerNorm over D=512 of (x + r), 256 threads per row
__global__ void ln_kernel(const float* __restrict__ x, const float* __restrict__ r,
                          const float* __restrict__ g, const float* __restrict__ b,
                          float* __restrict__ out)
{
  int row = blockIdx.x;
  int tid = threadIdx.x;
  const float* xr = x + (long)row * T_D;
  const float* rr = r ? r + (long)row * T_D : nullptr;
  float vals[2], s = 0.f, s2 = 0.f;
#pragma unroll
  for (int i = 0; i < 2; ++i) {
    int c = tid + i * 256;
    float v = xr[c] + (rr ? rr[c] : 0.f);
    vals[i] = v; s += v; s2 += v * v;
  }
  for (int o = 16; o > 0; o >>= 1) { s += __shfl_xor(s, o, 32); s2 += __shfl_xor(s2, o, 32); }
  __shared__ float sa[8], sb[8];
  if ((tid & 31) == 0) { sa[tid >> 5] = s; sb[tid >> 5] = s2; }
  __syncthreads();
  s = 0.f; s2 = 0.f;
  for (int i = 0; i < 8; ++i) { s += sa[i]; s2 += sb[i]; }
  float mean = s * (1.0f / T_D);
  float var  = s2 * (1.0f / T_D) - mean * mean;
  float rstd = rsqrtf(var + LN_EPS);
  float* o = out + (long)row * T_D;
#pragma unroll
  for (int i = 0; i < 2; ++i) {
    int c = tid + i * 256;
    o[c] = g[c] * (vals[i] - mean) * rstd + b[c];
  }
}

// Row softmax over S=512 scores -> f16 probabilities (causal mask optional)
__global__ void attn_softmax_kernel(const float* __restrict__ sc,
                                    _Float16* __restrict__ pr, int masked)
{
  int row = blockIdx.x;                 // over B*H*S
  int q = row % T_S;
  int valid = masked ? (q + 1) : T_S;
  const float* x = sc + (long)row * T_S;
  _Float16* y = pr + (long)row * T_S;
  int tid = threadIdx.x;
  float e[2], m = -3.0e38f;
#pragma unroll
  for (int i = 0; i < 2; ++i) {
    int c = tid + i * 256;
    float v = (c < valid) ? x[c] : -3.0e38f;
    e[i] = v; m = fmaxf(m, v);
  }
  for (int o = 16; o > 0; o >>= 1) m = fmaxf(m, __shfl_xor(m, o, 32));
  __shared__ float sm[8];
  if ((tid & 31) == 0) sm[tid >> 5] = m;
  __syncthreads();
  m = sm[0];
  for (int i = 1; i < 8; ++i) m = fmaxf(m, sm[i]);
  float s = 0.f;
#pragma unroll
  for (int i = 0; i < 2; ++i) {
    float v = (e[i] > -1.0e38f) ? __expf(e[i] - m) : 0.f;
    e[i] = v; s += v;
  }
  for (int o = 16; o > 0; o >>= 1) s += __shfl_xor(s, o, 32);
  __shared__ float ss[8];
  if ((tid & 31) == 0) ss[tid >> 5] = s;
  __syncthreads();
  s = 0.f;
  for (int i = 0; i < 8; ++i) s += ss[i];
  float inv = 1.0f / s;
#pragma unroll
  for (int i = 0; i < 2; ++i) {
    int c = tid + i * 256;
    y[c] = (_Float16)(e[i] * inv);
  }
}

// In-place softmax over the 32000-wide vocab dimension
__global__ void vocab_softmax_kernel(float* __restrict__ x)
{
  long row = blockIdx.x;                // over B*S
  float* p = x + row * (long)T_V;
  int tid = threadIdx.x;
  float m = -3.0e38f;
  for (int c = tid; c < T_V; c += 256) m = fmaxf(m, p[c]);
  for (int o = 16; o > 0; o >>= 1) m = fmaxf(m, __shfl_xor(m, o, 32));
  __shared__ float sm[8];
  if ((tid & 31) == 0) sm[tid >> 5] = m;
  __syncthreads();
  m = sm[0];
  for (int i = 1; i < 8; ++i) m = fmaxf(m, sm[i]);
  float s = 0.f;
  for (int c = tid; c < T_V; c += 256) s += __expf(p[c] - m);
  for (int o = 16; o > 0; o >>= 1) s += __shfl_xor(s, o, 32);
  __shared__ float ss[8];
  if ((tid & 31) == 0) ss[tid >> 5] = s;
  __syncthreads();
  s = 0.f;
  for (int i = 0; i < 8; ++i) s += ss[i];
  float inv = 1.0f / s;
  for (int c = tid; c < T_V; c += 256) p[c] = __expf(p[c] - m) * inv;
}

// ---------------------------------------------------------------------------
// Host orchestration
// ---------------------------------------------------------------------------
extern "C" void kernel_launch(void* const* d_in, const int* in_sizes, int n_in,
                              void* d_out, int out_size, void* d_ws, size_t ws_size,
                              hipStream_t stream)
{
  (void)in_sizes; (void)n_in; (void)out_size; (void)ws_size;

  const long long* src = (const long long*)d_in[0];
  const long long* dst = (const long long*)d_in[1];
  const float*     pos = (const float*)d_in[2];

  int p = 3;
  auto nf = [&]() -> const float* { return (const float*)d_in[p++]; };
  const float* enc_emb = nf();
  const float* dec_emb = nf();
  const float* out_W   = nf();
  const float* out_b   = nf();

  struct MHA  { const float *Wq, *bq, *Wk, *bk, *Wo, *bo; };
  struct LNP  { const float *g, *b; };
  struct FFNP { const float *W1, *b1, *W2, *b2; };
  struct EncL { MHA mha; LNP ln1; FFNP ffn; LNP ln2; };
  struct DecL { MHA s; LNP ln1; MHA c; LNP ln2; FFNP ffn; LNP ln3; };
  auto rdM = [&]() { MHA m; m.Wq=nf(); m.bq=nf(); m.Wk=nf(); m.bk=nf(); m.Wo=nf(); m.bo=nf(); return m; };
  auto rdL = [&]() { LNP l; l.g=nf(); l.b=nf(); return l; };
  auto rdF = [&]() { FFNP f; f.W1=nf(); f.b1=nf(); f.W2=nf(); f.b2=nf(); return f; };
  EncL enc[6]; DecL dec[6];
  for (int i = 0; i < 6; ++i) { enc[i].mha=rdM(); enc[i].ln1=rdL(); enc[i].ffn=rdF(); enc[i].ln2=rdL(); }
  for (int i = 0; i < 6; ++i) { dec[i].s=rdM(); dec[i].ln1=rdL(); dec[i].c=rdM(); dec[i].ln2=rdL(); dec[i].ffn=rdF(); dec[i].ln3=rdL(); }

  // --- workspace: only buffers live during the final logits GEMM go in d_ws ---
  char* wsp = (char*)d_ws; size_t woff = 0;
  auto walloc = [&](size_t bytes) { void* r = wsp + woff; woff += (bytes + 255) & ~(size_t)255; return r; };
  _Float16* abf  = (_Float16*)walloc((size_t)T_M * T_DFF * 2);  // f16 activation pack (8.4MB)
  _Float16* wtbf = (_Float16*)walloc((size_t)T_V * T_D * 2);    // f16 weight pack (32.8MB)

  // --- everything else is dead before the final GEMM: carve it out of d_out ---
  char* ob = (char*)d_out; size_t ooff = 0;
  auto oalloc = [&](size_t bytes) { void* r = ob + ooff; ooff += (bytes + 255) & ~(size_t)255; return r; };
  float*    scoresF = (float*)   oalloc((size_t)T_B * T_H * T_S * T_S * 4);  // also FFN hidden
  _Float16* probsH  = (_Float16*)oalloc((size_t)T_B * T_H * T_S * T_S * 2);
  float*    hEnc = (float*)oalloc((size_t)T_M * T_D * 4);
  float*    hDec = (float*)oalloc((size_t)T_M * T_D * 4);
  float*    ln1b = (float*)oalloc((size_t)T_M * T_D * 4);
  float*    ln2b = (float*)oalloc((size_t)T_M * T_D * 4);
  float*    tmpO = (float*)oalloc((size_t)T_M * T_D * 4);
  float*    qf   = (float*)oalloc((size_t)T_M * T_D * 4);      // also concat f32
  float*    kf_  = (float*)oalloc((size_t)T_M * T_D * 4);
  _Float16* kvbf  = (_Float16*)oalloc((size_t)T_M * T_D * 2);
  _Float16* qbf   = (_Float16*)oalloc((size_t)T_M * T_D * 2);
  _Float16* kbf   = (_Float16*)oalloc((size_t)T_M * T_D * 2);
  _Float16* catbf = (_Float16*)oalloc((size_t)T_M * T_D * 2);
  _Float16* vtbf  = (_Float16*)oalloc((size_t)T_B * T_H * T_DR * T_S * 2);
  float*    ffnF  = scoresF;  // alias: attention and FFN never overlap in time

  auto cvt = [&](const float* in, _Float16* o, long n) {
    unsigned blocks = (unsigned)((n + 255) / 256);
    cvt_f32_f16_kernel<<<dim3(blocks), 256, 0, stream>>>(in, o, n);
  };
  auto gemm = [&](const _Float16* A, long lda, const _Float16* Bt, long ldb,
                  float* C, long ldc, int M, int N, int K,
                  const float* bias, float scale, int relu,
                  int batch, int zdiv,
                  long aO, long aI, long bO, long bI, long cO, long cI) {
    dim3 g((N + BN - 1) / BN, (M + BM - 1) / BM, batch);
    gemm_f16_wmma<<<g, 256, 0, stream>>>(A, lda, Bt, ldb, C, ldc, M, N, K,
                                         bias, scale, relu, zdiv, aO, aI, bO, bI, cO, cI);
  };
  auto ln = [&](const float* x, const float* r, const LNP& pl, float* o) {
    ln_kernel<<<T_M, 256, 0, stream>>>(x, r, pl.g, pl.b, o);
  };

  auto mha = [&](const float* xq, const float* xkv, const MHA& pm, int masked, float* outp) {
    cvt(xq,  abf,  (long)T_M * T_D);
    cvt(xkv, kvbf, (long)T_M * T_D);
    // q = (x @ Wq + bq) / sqrt(DR)
    wt_heads_kernel<<<(T_H * T_DR * T_D) / 256, 256, 0, stream>>>(pm.Wq, wtbf);
    gemm(abf, T_D, wtbf, T_D, qf, T_D, T_M, T_D, T_D, pm.bq, 0.125f, 0, 1, 1, 0, 0, 0, 0, 0, 0);
    // k = xkv @ Wk + bk   (v = k, faithful to reference)
    wt_heads_kernel<<<(T_H * T_DR * T_D) / 256, 256, 0, stream>>>(pm.Wk, wtbf);
    gemm(kvbf, T_D, wtbf, T_D, kf_, T_D, T_M, T_D, T_D, pm.bk, 1.0f, 0, 1, 1, 0, 0, 0, 0, 0, 0);
    cvt(qf,  qbf, (long)T_M * T_D);
    cvt(kf_, kbf, (long)T_M * T_D);
    vt_pack_kernel<<<(T_B * T_H * T_DR * T_S) / 256, 256, 0, stream>>>(kf_, vtbf);
    // scores[b,h] = Q K^T  (batched over 32 (b,h) pairs; head-strided views)
    gemm(qbf, T_D, kbf, T_D, scoresF, T_S, T_S, T_S, T_DR, nullptr, 1.0f, 0,
         T_B * T_H, T_H,
         (long)T_S * T_D, T_DR,            // A: +b*S*512, +h*64
         (long)T_S * T_D, T_DR,            // B: same view of k
         (long)T_H * T_S * T_S, (long)T_S * T_S);
    attn_softmax_kernel<<<T_B * T_H * T_S, 256, 0, stream>>>(scoresF, probsH, masked);
    // o[b,h] = P V   -> written straight into concat layout [M, H*DR]
    gemm(probsH, T_S, vtbf, T_S, qf, T_D, T_S, T_DR, T_S, nullptr, 1.0f, 0,
         T_B * T_H, T_H,
         (long)T_H * T_S * T_S, (long)T_S * T_S,
         (long)T_H * T_DR * T_S, (long)T_DR * T_S,
         (long)T_S * T_D, T_DR);
    cvt(qf, catbf, (long)T_M * T_D);
    wt_transpose_kernel<<<(T_D * T_D) / 256, 256, 0, stream>>>(pm.Wo, wtbf, T_D, T_D);
    gemm(catbf, T_D, wtbf, T_D, outp, T_D, T_M, T_D, T_D, pm.bo, 1.0f, 0, 1, 1, 0, 0, 0, 0, 0, 0);
  };

  auto ffn = [&](const float* x, const FFNP& pf, float* outp) {
    cvt(x, abf, (long)T_M * T_D);
    wt_transpose_kernel<<<(T_D * T_DFF) / 256, 256, 0, stream>>>(pf.W1, wtbf, T_D, T_DFF);
    gemm(abf, T_D, wtbf, T_D, ffnF, T_DFF, T_M, T_DFF, T_D, pf.b1, 1.0f, 1, 1, 1, 0, 0, 0, 0, 0, 0);
    cvt(ffnF, abf, (long)T_M * T_DFF);
    wt_transpose_kernel<<<(T_DFF * T_D) / 256, 256, 0, stream>>>(pf.W2, wtbf, T_DFF, T_D);
    gemm(abf, T_DFF, wtbf, T_DFF, outp, T_D, T_M, T_D, T_DFF, pf.b2, 1.0f, 0, 1, 1, 0, 0, 0, 0, 0, 0);
  };

  // ---------------- encoder ----------------
  embed_kernel<<<T_M, 256, 0, stream>>>(src, enc_emb, pos, hEnc);
  for (int l = 0; l < 6; ++l) {
    mha(hEnc, hEnc, enc[l].mha, 0, tmpO);
    ln(hEnc, tmpO, enc[l].ln1, ln1b);
    ffn(ln1b, enc[l].ffn, tmpO);
    ln(tmpO, ln1b, enc[l].ln2, hEnc);
  }

  // ---------------- decoder ----------------
  embed_kernel<<<T_M, 256, 0, stream>>>(dst, dec_emb, pos, hDec);
  for (int l = 0; l < 6; ++l) {
    mha(hDec, hDec, dec[l].s, 1, tmpO);          // masked self-attention
    ln(hDec, tmpO, dec[l].ln1, ln1b);
    mha(ln1b, hEnc, dec[l].c, 0, tmpO);          // cross-attention
    ln(ln1b, tmpO, dec[l].ln2, ln2b);
    ffn(ln1b, dec[l].ffn, tmpO);                 // faithful quirk: FFN(ln1)
    ln(tmpO, ln2b, dec[l].ln3, hDec);            // residual added to ln2
  }

  // ---------------- logits + vocab softmax ----------------
  cvt(hDec, abf, (long)T_M * T_D);
  wt_transpose_kernel<<<(unsigned)(((long)T_D * T_V) / 256), 256, 0, stream>>>(out_W, wtbf, T_D, T_V);
  gemm(abf, T_D, wtbf, T_D, (float*)d_out, T_V, T_M, T_V, T_D, out_b, 1.0f, 0,
       1, 1, 0, 0, 0, 0, 0, 0);
  vocab_softmax_kernel<<<T_M, 256, 0, stream>>>((float*)d_out);
}
